// MultiHeadAttention_39402029973654
// MI455X (gfx1250) — matrix operand order repaired and checked by reading, not compiled, not run on hardware
//
#include <hip/hip_runtime.h>

// ---------------------------------------------------------------------------
// MHA on gfx1250 (MI455X): bf16 WMMA everywhere, flash-attention middle.
// B=2, S=4096, D=1024, H=16, DK=64.
//
// Pipeline (all on `stream`):
//   1) fp32 -> bf16 conversion of Q,K,V and Wq,Wk,Wv,Wo (one pass, HBM-bound)
//   2) 3 projection GEMMs (bf16 WMMA, 64x64 tile per wave, fp32 accum)
//        q,k stored head-major [b][h][s][dk]; v stored transposed+column-
//        permuted [b][h][dk][s'] so the PV B-operand load stays contiguous
//        while the P tile packs into b32 LDS stores.
//   3) flash attention (online softmax, per-wave 16 query rows, T in 32-steps)
//   4) output GEMM -> fp32 d_out
// ---------------------------------------------------------------------------

typedef __bf16 bf16;
typedef __attribute__((ext_vector_type(16))) __bf16 v16bf;
typedef __attribute__((ext_vector_type(8)))  __bf16 v8bf;
typedef __attribute__((ext_vector_type(8)))  float  v8f;
typedef __attribute__((ext_vector_type(4)))  float  v4f;

union V16U { v16bf v; v8bf h[2]; };
union PKU  { bf16 b[2]; unsigned u; };

#define WMMA_BF16(A, Bv, C) \
  __builtin_amdgcn_wmma_f32_16x16x32_bf16(false, (A), false, (Bv), (short)0, (C), false, false)

__device__ inline v8f vzero8() {
  v8f z;
#pragma unroll
  for (int i = 0; i < 8; ++i) z[i] = 0.0f;
  return z;
}

__device__ inline v8bf cvt8(v4f a, v4f b) {
  v8bf r;
  r[0] = (bf16)a[0]; r[1] = (bf16)a[1]; r[2] = (bf16)a[2]; r[3] = (bf16)a[3];
  r[4] = (bf16)b[0]; r[5] = (bf16)b[1]; r[6] = (bf16)b[2]; r[7] = (bf16)b[3];
  return r;
}

// A-fragment (16x32 bf16, MxK): lane l holds row m = base + (l&15).
// half = l>>4:  elems 0..7  -> K = kb + half*8 + 0..7
//               elems 8..15 -> K = kb + 16 + half*8 + 0..7
__device__ inline v16bf loadA_bf(const bf16* X, int ld, int m, int kb, int half) {
  const bf16* p = X + (size_t)m * ld + kb + half * 8;
  V16U u;
  u.h[0] = *(const v8bf*)p;
  u.h[1] = *(const v8bf*)(p + 16);
  return u.v;
}

// B-fragment (32x16 bf16, KxN) for C = X * W^T:  B[k][n] = W[n][k].
// lane l holds column n = base + (l&15); elems 0..7 -> K = kb + half*8 + 0..7,
// elems 8..15 -> K = kb + 16 + half*8 + 0..7 (same order as loadA_bf).
__device__ inline v16bf loadBt_bf(const bf16* W, int ld, int n, int kb, int half) {
  const bf16* p = W + (size_t)n * ld + kb + half * 8;
  V16U u;
  u.h[0] = *(const v8bf*)p;
  u.h[1] = *(const v8bf*)(p + 16);
  return u.v;
}

// ---------------------------------------------------------------------------
// fp32 -> bf16 conversion, 8 elements per thread (b128 in, b128/2 out).
// ---------------------------------------------------------------------------
__global__ __launch_bounds__(256) void cvt_bf16(const float* __restrict__ src,
                                                void* __restrict__ dstv, int n8) {
  const int i = blockIdx.x * 256 + threadIdx.x;
  if (i >= n8) return;
  const v4f a = *(const v4f*)(src + (size_t)i * 8);
  const v4f b = *(const v4f*)(src + (size_t)i * 8 + 4);
  ((v8bf*)dstv)[i] = cvt8(a, b);
}

// ---------------------------------------------------------------------------
// GEMM: C[M=8192][N=1024] = X[8192][1024] @ W[1024][1024]^T, all bf16 in.
// One wave computes a 64x64 tile (4x4 accumulators, 16 WMMA per 32-K step).
// SMODE 0: bf16 store, head-major  dst[b][h][s][dk]            (q, k)
// SMODE 1: bf16 store, transposed+permuted dst[b][h][dk][s']   (v)
//          s' permutes within 32-blocks: sigma(s) pairs P columns for b32
//          LDS packing in the attention kernel.
// SMODE 2: f32  store, row-major   dst[m][n]                   (final out)
// ---------------------------------------------------------------------------
template <int SMODE>
__global__ __launch_bounds__(128) void mha_gemm(const void* __restrict__ Xv,
                                                const void* __restrict__ Wv,
                                                void* __restrict__ dstv) {
  constexpr int KD = 1024;
  constexpr int ND = 1024;
  const bf16* X = (const bf16*)Xv;
  const bf16* W = (const bf16*)Wv;
  const int lane = threadIdx.x & 31;
  const int wid  = threadIdx.x >> 5;
  const int half = lane >> 4;
  const int ln   = lane & 15;
  const int w    = blockIdx.x * 4 + wid;
  const int tN   = (w & 15) * 64;   // N/64 = 16 tiles
  const int tM   = (w >> 4) * 64;   // M/64 = 128 tiles

  v8f acc[4][4];
#pragma unroll
  for (int mi = 0; mi < 4; ++mi)
#pragma unroll
    for (int ni = 0; ni < 4; ++ni) acc[mi][ni] = vzero8();

  for (int kb = 0; kb < KD; kb += 32) {
    v16bf Af[4], Bf[4];
#pragma unroll
    for (int mi = 0; mi < 4; ++mi)
      Af[mi] = loadA_bf(X, KD, tM + mi * 16 + ln, kb, half);
#pragma unroll
    for (int ni = 0; ni < 4; ++ni)
      Bf[ni] = loadBt_bf(W, KD, tN + ni * 16 + ln, kb, half);
#pragma unroll
    for (int mi = 0; mi < 4; ++mi)
#pragma unroll
      for (int ni = 0; ni < 4; ++ni)
        acc[mi][ni] = WMMA_BF16(Af[mi], Bf[ni], acc[mi][ni]);
  }

  // Epilogue. C layout: elem r -> M = r + half*8, N = ln.
#pragma unroll
  for (int mi = 0; mi < 4; ++mi) {
#pragma unroll
    for (int ni = 0; ni < 4; ++ni) {
#pragma unroll
      for (int r = 0; r < 8; ++r) {
        const int m = tM + mi * 16 + half * 8 + r;
        const int n = tN + ni * 16 + ln;
        const float val = acc[mi][ni][r];
        if (SMODE == 2) {
          ((float*)dstv)[(size_t)m * ND + n] = val;
        } else {
          const int b = m >> 12, s = m & 4095;   // S = 4096
          const int h = n >> 6,  dk = n & 63;    // DK = 64
          bf16* dst = (bf16*)dstv;
          if (SMODE == 0) {
            dst[(((size_t)(b * 16 + h) * 4096 + s) << 6) + dk] = (bf16)val;
          } else {
            // sigma within 32-token blocks: s' = 2*(s&15) + ((s>>4)&1)
            const int sp = (s & ~31) | ((s & 15) << 1) | ((s >> 4) & 1);
            dst[(((size_t)(b * 16 + h) * 64 + dk) << 12) + sp] = (bf16)val;
          }
        }
      }
    }
  }
}

// ---------------------------------------------------------------------------
// Flash attention, non-causal. One wave owns 16 query rows x DK=64,
// streams T in steps of 32 with online softmax.
//  qh, kh : bf16 [b][h][4096][64]   vT : bf16 [b][h][64][4096] (sigma-permuted)
//  hb     : bf16 [b][s][h*64+dk]  (concat heads, row-major for final GEMM)
// ---------------------------------------------------------------------------
__global__ __launch_bounds__(256) void mha_attn(const void* __restrict__ qhv,
                                                const void* __restrict__ khv,
                                                const void* __restrict__ vtv,
                                                void* __restrict__ hbv) {
  __shared__ alignas(16) bf16 plds[8][16 * 32];   // per-wave 16x32 P tile

  const bf16* qh = (const bf16*)qhv;
  const bf16* kh = (const bf16*)khv;
  const bf16* vT = (const bf16*)vtv;
  bf16*       hb = (bf16*)hbv;

  const int lane = threadIdx.x & 31;
  const int wid  = threadIdx.x >> 5;
  const int half = lane >> 4;
  const int ln   = lane & 15;
  const int w    = blockIdx.x * 8 + wid;
  const int mt   = w & 255;          // S/16 = 256 query tiles
  const int h    = (w >> 8) & 15;
  const int b    = w >> 12;

  const size_t headoff = (size_t)(b * 16 + h) * 4096 * 64;
  const bf16* qb = qh + headoff;
  const bf16* kb = kh + headoff;
  const bf16* vb = vT + headoff;     // [64][4096]

  // Preload Q A-fragments (16x64 = two 16x32 k-steps).
  v16bf aq[2];
  {
    const bf16* qrow = qb + (size_t)(mt * 16 + ln) * 64;
#pragma unroll
    for (int kk = 0; kk < 2; ++kk) {
      V16U u;
      u.h[0] = *(const v8bf*)(qrow + kk * 32 + half * 8);
      u.h[1] = *(const v8bf*)(qrow + kk * 32 + 16 + half * 8);
      aq[kk] = u.v;
    }
  }

  float mrow[8], lrow[8];
#pragma unroll
  for (int r = 0; r < 8; ++r) { mrow[r] = -3.0e38f; lrow[r] = 0.0f; }
  v8f acc_o[4];
#pragma unroll
  for (int ni = 0; ni < 4; ++ni) acc_o[ni] = vzero8();

  bf16* pl = plds[wid];
  unsigned* pl32 = (unsigned*)pl;

  for (int t = 0; t < 4096; t += 32) {
    // ---- S = Q K^T for 32 key columns (two 16x16 tiles side by side) ----
    v8f s0 = vzero8(), s1 = vzero8();
#pragma unroll
    for (int kk = 0; kk < 2; ++kk) {
      v16bf bk0 = loadBt_bf(kb, 64, t + ln,      kk * 32, half);
      v16bf bk1 = loadBt_bf(kb, 64, t + 16 + ln, kk * 32, half);
      s0 = WMMA_BF16(aq[kk], bk0, s0);
      s1 = WMMA_BF16(aq[kk], bk1, s1);
    }

    // ---- online softmax (row = half*8 + r lives on the 16 lanes of a half) --
#pragma unroll
    for (int r = 0; r < 8; ++r) {
      const float x0 = s0[r] * 0.125f;   // 1/sqrt(64)
      const float x1 = s1[r] * 0.125f;
      float mx = fmaxf(x0, x1);
      mx = fmaxf(mx, __shfl_xor(mx, 1));
      mx = fmaxf(mx, __shfl_xor(mx, 2));
      mx = fmaxf(mx, __shfl_xor(mx, 4));
      mx = fmaxf(mx, __shfl_xor(mx, 8));
      const float mnew = fmaxf(mrow[r], mx);
      const float al   = __expf(mrow[r] - mnew);
      const float e0   = __expf(x0 - mnew);
      const float e1   = __expf(x1 - mnew);
      float rs = e0 + e1;
      rs += __shfl_xor(rs, 1);
      rs += __shfl_xor(rs, 2);
      rs += __shfl_xor(rs, 4);
      rs += __shfl_xor(rs, 8);
      lrow[r] = lrow[r] * al + rs;
      mrow[r] = mnew;
#pragma unroll
      for (int ni = 0; ni < 4; ++ni) acc_o[ni][r] *= al;
      // sigma-packed P store: columns 2*ln (orig ln) and 2*ln+1 (orig 16+ln)
      PKU pk;
      pk.b[0] = (bf16)e0;
      pk.b[1] = (bf16)e1;
      pl32[(half * 8 + r) * 16 + ln] = pk.u;
    }

    // Wave-local LDS RAW: wait for the DS stores, then read A-layout.
    asm volatile("s_wait_dscnt 0" ::: "memory");
    v16bf ap;
    {
      V16U u;
      u.h[0] = *(const v8bf*)(pl + ln * 32 + half * 8);
      u.h[1] = *(const v8bf*)(pl + ln * 32 + 16 + half * 8);
      ap = u.v;
    }

    // ---- O += P V  (K-dim = 32 sigma-ordered key rows; vT rows contiguous) --
#pragma unroll
    for (int ni = 0; ni < 4; ++ni) {
      const bf16* vrow = vb + (size_t)(ni * 16 + ln) * 4096 + t + half * 8;
      V16U uv;
      uv.h[0] = *(const v8bf*)vrow;
      uv.h[1] = *(const v8bf*)(vrow + 16);
      acc_o[ni] = WMMA_BF16(ap, uv.v, acc_o[ni]);
    }
  }

  // ---- normalize and store heads (concat layout for the final GEMM) ----
#pragma unroll
  for (int ni = 0; ni < 4; ++ni) {
#pragma unroll
    for (int r = 0; r < 8; ++r) {
      const int srow = mt * 16 + half * 8 + r;
      const int col  = h * 64 + ni * 16 + ln;
      const float o  = acc_o[ni][r] / lrow[r];
      hb[((size_t)(b * 4096 + srow)) * 1024 + col] = (bf16)o;
    }
  }
}

// ---------------------------------------------------------------------------
extern "C" void kernel_launch(void* const* d_in, const int* in_sizes, int n_in,
                              void* d_out, int out_size, void* d_ws, size_t ws_size,
                              hipStream_t stream) {
  (void)in_sizes; (void)n_in; (void)out_size; (void)ws_size;

  const float* Q  = (const float*)d_in[0];
  const float* Kx = (const float*)d_in[1];
  const float* V  = (const float*)d_in[2];
  const float* Wq = (const float*)d_in[3];
  const float* Wk = (const float*)d_in[4];
  const float* Wv = (const float*)d_in[5];
  const float* Wo = (const float*)d_in[6];

  char* ws = (char*)d_ws;
  const size_t SZX = (size_t)2 * 4096 * 1024 * 2;   // 16 MiB (8M bf16)
  const size_t SZW = (size_t)1024 * 1024 * 2;       //  2 MiB (1M bf16)
  void* Qbf = (void*)(ws + 0 * SZX);
  void* Kbf = (void*)(ws + 1 * SZX);
  void* Vbf = (void*)(ws + 2 * SZX);
  char* wb  = ws + 3 * SZX;
  void* Wqb = (void*)(wb + 0 * SZW);
  void* Wkb = (void*)(wb + 1 * SZW);
  void* Wvb = (void*)(wb + 2 * SZW);
  void* Wob = (void*)(wb + 3 * SZW);
  char* hbuf = wb + 4 * SZW;
  void* qhp = (void*)(hbuf + 0 * SZX);
  void* khp = (void*)(hbuf + 1 * SZX);
  void* vtp = (void*)(hbuf + 2 * SZX);
  void* hbp = Qbf;   // Q bf16 copy is dead after the first GEMM; reuse as heads

  // 1) conversions (8M elems -> 1M threads; 1M elems -> 128K threads)
  const int nX8 = 2 * 4096 * 1024 / 8, nW8 = 1024 * 1024 / 8;
  cvt_bf16<<<dim3(nX8 / 256), dim3(256), 0, stream>>>(Q,  Qbf, nX8);
  cvt_bf16<<<dim3(nX8 / 256), dim3(256), 0, stream>>>(Kx, Kbf, nX8);
  cvt_bf16<<<dim3(nX8 / 256), dim3(256), 0, stream>>>(V,  Vbf, nX8);
  cvt_bf16<<<dim3(nW8 / 256), dim3(256), 0, stream>>>(Wq, Wqb, nW8);
  cvt_bf16<<<dim3(nW8 / 256), dim3(256), 0, stream>>>(Wk, Wkb, nW8);
  cvt_bf16<<<dim3(nW8 / 256), dim3(256), 0, stream>>>(Wv, Wvb, nW8);
  cvt_bf16<<<dim3(nW8 / 256), dim3(256), 0, stream>>>(Wo, Wob, nW8);

  // 2) projections: 2048 waves = 128x16 tiles of 64x64
  const dim3 gblk(128), ggrid(512);
  mha_gemm<0><<<ggrid, gblk, 0, stream>>>(Qbf, Wqb, qhp);
  mha_gemm<0><<<ggrid, gblk, 0, stream>>>(Kbf, Wkb, khp);
  mha_gemm<1><<<ggrid, gblk, 0, stream>>>(Vbf, Wvb, vtp);

  // 3) attention: B*H*(S/16) = 8192 waves
  mha_attn<<<dim3(1024), dim3(256), 0, stream>>>(qhp, khp, vtp, hbp);

  // 4) output projection -> fp32 d_out
  mha_gemm<2><<<ggrid, gblk, 0, stream>>>(hbp, Wob, d_out);
}
